// GINEConv_GNNB_3092376453269
// MI455X (gfx1250) — matrix operand order repaired and checked by reading, not compiled, not run on hardware
//
#include <hip/hip_runtime.h>
#include <hip/hip_bf16.h>

typedef _Float16 v16h __attribute__((ext_vector_type(16)));
typedef _Float16 h8v  __attribute__((ext_vector_type(8)));
typedef float    v8f  __attribute__((ext_vector_type(8)));
typedef float    v4f  __attribute__((ext_vector_type(4)));

#define DD 64
#define HH 128

union U16H { v16h v; h8v h[2]; };

// ---- fragment helpers -------------------------------------------------------

// B fragment (f16, 32x16 tile): lane n=lr, this lane needs K values p[0..7]
// and p[16..23] (p already offset by n*stride + kstart + 8*half). 16B-aligned.
__device__ __forceinline__ v16h load_b_frag(const _Float16* __restrict__ p) {
    U16H u;
    u.h[0] = *(const h8v*)(p);
    u.h[1] = *(const h8v*)(p + 16);
    return u.v;
}

// A fragment (f16, 16x32 tile) built from fp32 row data:
// elements 0..7  <- p0[0..7]   (K = koff .. koff+7)
// elements 8..15 <- p1[0..7]   (K = koff+16 .. koff+23)
__device__ __forceinline__ v16h cvt_a_frag(const float* __restrict__ p0,
                                           const float* __restrict__ p1) {
    v4f a0 = *(const v4f*)(p0);
    v4f a1 = *(const v4f*)(p0 + 4);
    v4f b0 = *(const v4f*)(p1);
    v4f b1 = *(const v4f*)(p1 + 4);
    v16h r;
#pragma unroll
    for (int i = 0; i < 4; ++i) {
        r[i]      = (_Float16)a0[i];
        r[i + 4]  = (_Float16)a1[i];
        r[i + 8]  = (_Float16)b0[i];
        r[i + 12] = (_Float16)b1[i];
    }
    return r;
}

// A fragment straight from f16 LDS row data (chunks already f16).
__device__ __forceinline__ v16h load_a_frag_h(const _Float16* __restrict__ p0,
                                              const _Float16* __restrict__ p1) {
    U16H u;
    u.h[0] = *(const h8v*)(p0);
    u.h[1] = *(const h8v*)(p1);
    return u.v;
}

#define WMMA_F16(A, B, C) \
    __builtin_amdgcn_wmma_f32_16x16x32_f16(false, (A), false, (B), (short)0, (C), false, false)

// ---- kernel 0: weight fp32 -> f16 ------------------------------------------

__global__ __launch_bounds__(256) void prep_weights(
    const float* __restrict__ lin_w, const float* __restrict__ w0,
    const float* __restrict__ w1, _Float16* __restrict__ lwf,
    _Float16* __restrict__ w0f, _Float16* __restrict__ w1f) {
    int i = blockIdx.x * blockDim.x + threadIdx.x;
    if (i < DD * DD) lwf[i] = (_Float16)lin_w[i];
    if (i < HH * DD) w0f[i] = (_Float16)w0[i];
    if (i < DD * HH) w1f[i] = (_Float16)w1[i];
}

// ---- kernel 1: edge pass ----------------------------------------------------
// Per wave: 16 edges. e = edge_attr @ lin_w^T (+b), msg = relu(x[src]+e),
// atomic scatter-add into agg (pre-initialized with x).

__global__ __launch_bounds__(256) void edge_kernel(
    const float* __restrict__ x, const long long* __restrict__ ei,
    const float* __restrict__ edge_attr, const _Float16* __restrict__ lwf,
    const float* __restrict__ lin_b, float* __restrict__ agg, int E) {
    long long wave = ((long long)blockIdx.x * blockDim.x + threadIdx.x) >> 5;
    long long e0 = wave * 16;
    if (e0 >= E) return;               // wave-uniform: EXEC stays all-ones

    int l    = threadIdx.x & 31;
    int hh   = l >> 4;                 // lane half
    int lr   = l & 15;
    int koff = hh * 8;                 // A-fragment K chunk base
    int kb   = hh * 16;                // unused; B fragments also use koff

    // A fragments: rows = edges e0..e0+15, fp32 -> f16 on the fly
    long long arow = e0 + lr;
    if (arow >= E) arow = E - 1;
    const float* ap = edge_attr + arow * DD;
    v16h a0 = cvt_a_frag(ap + koff,      ap + 16 + koff);   // K 0..31
    v16h a1 = cvt_a_frag(ap + 32 + koff, ap + 48 + koff);   // K 32..63

    v8f c[4];
#pragma unroll
    for (int t = 0; t < 4; ++t) {
        int n = t * 16 + lr;
        const _Float16* wp = lwf + n * DD;   // B[k,n] = lin_w[n,k]
        v16h bq0 = load_b_frag(wp + koff);
        v16h bq1 = load_b_frag(wp + 32 + koff);
        v8f cc = {};
        cc = WMMA_F16(a0, bq0, cc);
        cc = WMMA_F16(a1, bq1, cc);
        c[t] = cc;
    }

    // epilogue: bias + gather x[src] + relu + atomic scatter to agg[dst]
#pragma unroll
    for (int t = 0; t < 4; ++t) {
        int n = t * 16 + lr;
        float bias = lin_b[n];
#pragma unroll
        for (int v = 0; v < 8; ++v) {
            long long e = e0 + v + hh * 8;   // C layout: row = v + 8*half
            if (e < E) {
                long long s = ei[e];
                long long d = ei[(long long)E + e];
                float val = c[t][v] + bias + x[s * DD + n];
                val = fmaxf(val, 0.0f);
                unsafeAtomicAdd(&agg[d * DD + n], val);
            }
        }
    }
}

// ---- kernel 2: fused node MLP ----------------------------------------------
// Per wave: 16 nodes. h = agg (already x + scatter-sum).
// h1 = relu(h @ w0^T + b0)  [16x128], staged via LDS for layout transpose,
// out = h1 @ w1^T + b1      [16x64].

__global__ __launch_bounds__(256) void node_kernel(
    const float* __restrict__ agg, const _Float16* __restrict__ w0f,
    const float* __restrict__ bias0, const _Float16* __restrict__ w1f,
    const float* __restrict__ bias1, float* __restrict__ out, int N) {
    __shared__ _Float16 lds[8][16 * HH];   // 4 KB per wave, 32 KB per block

    int wib = threadIdx.x >> 5;
    long long wave  = ((long long)blockIdx.x * blockDim.x + threadIdx.x) >> 5;
    long long tiles = ((long long)N + 15) >> 4;
    bool active = wave < tiles;
    long long node0 = (active ? wave : 0) * 16;   // clamp: keep barriers uniform

    int l    = threadIdx.x & 31;
    int hh   = l >> 4;
    int lr   = l & 15;
    int koff = hh * 8;
    int kb   = hh * 16;

    // GEMM1: A from agg (fp32 -> f16)
    long long arow = node0 + lr;
    if (arow >= N) arow = N - 1;
    const float* ap = agg + arow * DD;
    v16h a0 = cvt_a_frag(ap + koff,      ap + 16 + koff);
    v16h a1 = cvt_a_frag(ap + 32 + koff, ap + 48 + koff);

    _Float16* my = lds[wib];
#pragma unroll
    for (int t = 0; t < 8; ++t) {          // 8 output col-tiles -> HID=128
        int n = t * 16 + lr;
        const _Float16* wp = w0f + n * DD; // B[k,n] = w0[n,k]
        v16h bq0 = load_b_frag(wp + koff);
        v16h bq1 = load_b_frag(wp + 32 + koff);
        v8f cc = {};
        cc = WMMA_F16(a0, bq0, cc);
        cc = WMMA_F16(a1, bq1, cc);
        float bb = bias0[n];
#pragma unroll
        for (int v = 0; v < 8; ++v) {      // relu + cvt -> LDS row-major
            int m = v + hh * 8;
            my[m * HH + n] = (_Float16)fmaxf(cc[v] + bb, 0.0f);
        }
    }

    __syncthreads();   // uniform across block (tiles clamped, no early return)

    // GEMM2: A from LDS h1 tile (f16, row-major 16x128), K = 128 -> 4 steps
    v16h A[4];
#pragma unroll
    for (int q = 0; q < 4; ++q) {
        const _Float16* r = my + lr * HH + 32 * q;
        A[q] = load_a_frag_h(r + koff, r + 16 + koff);
    }

    v8f c2[4];
#pragma unroll
    for (int t = 0; t < 4; ++t) {          // 4 output col-tiles -> D=64
        int n = t * 16 + lr;
        const _Float16* wp = w1f + n * HH; // B[k,n] = w1[n,k]
        v8f cc = {};
#pragma unroll
        for (int q = 0; q < 4; ++q) {
            v16h bq = load_b_frag(wp + 32 * q + koff);
            cc = WMMA_F16(A[q], bq, cc);
        }
        c2[t] = cc;
    }

    if (!active) return;
#pragma unroll
    for (int t = 0; t < 4; ++t) {
        int n = t * 16 + lr;
        float bb = bias1[n];
#pragma unroll
        for (int v = 0; v < 8; ++v) {
            int m = v + hh * 8;
            long long node = node0 + m;
            if (node < N) out[node * DD + n] = c2[t][v] + bb;
        }
    }
}

// ---- launch -----------------------------------------------------------------

extern "C" void kernel_launch(void* const* d_in, const int* in_sizes, int n_in,
                              void* d_out, int out_size, void* d_ws, size_t ws_size,
                              hipStream_t stream) {
    const float*     x         = (const float*)d_in[0];
    const long long* ei        = (const long long*)d_in[1];   // int64 [2,E]
    const float*     edge_attr = (const float*)d_in[2];
    const float*     lin_w     = (const float*)d_in[3];
    const float*     lin_b     = (const float*)d_in[4];
    const float*     w0        = (const float*)d_in[5];
    const float*     b0        = (const float*)d_in[6];
    const float*     w1        = (const float*)d_in[7];
    const float*     b1        = (const float*)d_in[8];

    int N = in_sizes[0] / DD;        // 100000
    int E = in_sizes[2] / DD;        // 1600000

    // workspace: [agg fp32 N*64][lin_w f16][w0 f16][w1 f16]
    float* agg = (float*)d_ws;
    size_t aggBytes = (size_t)N * DD * sizeof(float);
    size_t off = (aggBytes + 255) & ~(size_t)255;
    _Float16* lwf = (_Float16*)((char*)d_ws + off);
    _Float16* w0f = lwf + DD * DD;
    _Float16* w1f = w0f + HH * DD;

    prep_weights<<<(HH * DD + 255) / 256, 256, 0, stream>>>(lin_w, w0, w1,
                                                            lwf, w0f, w1f);
    // agg = x  (folds the '(1+eps)*x + agg' term; eps = 0)
    hipMemcpyAsync(agg, x, aggBytes, hipMemcpyDeviceToDevice, stream);

    int edgeBlocks = (E + 127) / 128;          // 8 waves/block, 16 edges/wave
    edge_kernel<<<edgeBlocks, 256, 0, stream>>>(x, ei, edge_attr, lwf, lin_b,
                                                agg, E);

    int tiles = (N + 15) / 16;
    int nodeBlocks = (tiles + 7) / 8;          // 8 waves/block, 16 nodes/wave
    node_kernel<<<nodeBlocks, 256, 0, stream>>>(agg, w0f, b0, w1f, b1,
                                                (float*)d_out, N);
}
